// RNN_57621281243245
// MI455X (gfx1250) — matrix operand order repaired and checked by reading, not compiled, not run on hardware
//
#include <hip/hip_runtime.h>
#include <cmath>

#define SEQ     512
#define BATCH   64
#define INDIM   128
#define HID     512

#define RWG     4            // recurrence workgroups (one WGP each), N-slices of HID
#define RNS     (HID / RWG)  // 128 output columns per WG
#define WLD     520          // LDS row stride (f16) for W slice  [128][520] -> 130KB
#define HLD     520          // LDS row stride (f16) for staged h [ 64][520] ->  65KB

#define GBM     64
#define GBN     128
#define GBK     32
#define GLD     40           // LDS row stride (f16) for GEMM staging (16B aligned, conflict-free)

typedef __attribute__((ext_vector_type(16))) _Float16 v16h;
typedef __attribute__((ext_vector_type(8)))  _Float16 v8h;
typedef __attribute__((ext_vector_type(8)))  float    v8f;
typedef __attribute__((ext_vector_type(4)))  unsigned int u32x4;
typedef __attribute__((ext_vector_type(8)))  int          i32x8;
typedef __attribute__((ext_vector_type(4)))  int          i32x4;

#if __has_builtin(__builtin_amdgcn_tensor_load_to_lds) && \
    __has_builtin(__builtin_amdgcn_s_wait_tensorcnt)
#define HAVE_TDM 1
#else
#define HAVE_TDM 0
#endif

__device__ __forceinline__ v8f wmma_f32_f16(v16h a, v16h b, v8f c) {
  // D = A(16x32 f16) * B(32x16 f16) + C(16x16 f32)
  return __builtin_amdgcn_wmma_f32_16x16x32_f16(false, a, false, b, (short)0, c,
                                                false, false);
}

// A fragment (16x32 f16), ISA 7.12.2: lanes 0-15 row=lane, K = {0..7, 16..23};
// lanes 16-31 row=lane-16, K = {8..15, 24..31}.
__device__ __forceinline__ v16h load_a_frag(const _Float16* base, int row0, int ld,
                                            int k0, int lane) {
  const int r  = lane & 15;
  const int k1 = (lane >> 4) << 3;     // 0 or 8
  const _Float16* p = base + (row0 + r) * ld + k0 + k1;
  v8h lo = *(const v8h*)(p);           // K = k1 .. k1+7
  v8h hi = *(const v8h*)(p + 16);      // K = k1+16 .. k1+23
  return __builtin_shufflevector(lo, hi, 0,1,2,3,4,5,6,7,8,9,10,11,12,13,14,15);
}

// B fragment (32x16 f16), weights stored LDS[n][k]: lanes 0-15 col=lane, K = {0..7, 16..23};
// lanes 16-31 col=lane-16, K = {8..15, 24..31}.
__device__ __forceinline__ v16h load_b_frag(const _Float16* base, int col0, int ld,
                                            int k0, int lane) {
  const int c  = lane & 15;
  const int kb = (lane >> 4) << 3;     // 0 or 8
  const _Float16* p = base + (col0 + c) * ld + k0 + kb;
  v8h lo = *(const v8h*)(p);
  v8h hi = *(const v8h*)(p + 16);
  return __builtin_shufflevector(lo, hi, 0,1,2,3,4,5,6,7,8,9,10,11,12,13,14,15);
}

// ---------------------------------------------------------------------------
// Input-projection GEMM: out[m,n] = b_ih[n] + b_hh[n] + sum_k X[m,k] * W[n,k]
// M = SEQ*BATCH, N = HID, K in {INDIM, HID}. f16 operands, f32 accumulate.
// ---------------------------------------------------------------------------
__global__ __launch_bounds__(256)
void rnn_xw_gemm(const float* __restrict__ X, const float* __restrict__ W,
                 const float* __restrict__ bih, const float* __restrict__ bhh,
                 float* __restrict__ out, int K) {
  __shared__ _Float16 Ash[GBM * GLD];
  __shared__ _Float16 Bsh[GBN * GLD];

  const int tid  = threadIdx.x;
  const int lane = tid & 31;
  const int wave = tid >> 5;
  const int m0 = blockIdx.x * GBM;
  const int n0 = blockIdx.y * GBN;
  const int mw = (wave & 1) * 32;      // wave's 32-row slice inside block
  const int nw = (wave >> 1) * 32;     // wave's 32-col slice inside block

  v8f acc[2][2] = {};

  for (int k0 = 0; k0 < K; k0 += GBK) {
    __syncthreads();
    { // stage A: 64 x 32 f32 -> f16 (8 elems / thread)
      const int r  = tid >> 2;
      const int kk = (tid & 3) * 8;
      const float* src = X + (size_t)(m0 + r) * K + k0 + kk;
      float4 f0 = *(const float4*)(src);
      float4 f1 = *(const float4*)(src + 4);
      _Float16* dst = Ash + r * GLD + kk;
      dst[0]=(_Float16)f0.x; dst[1]=(_Float16)f0.y; dst[2]=(_Float16)f0.z; dst[3]=(_Float16)f0.w;
      dst[4]=(_Float16)f1.x; dst[5]=(_Float16)f1.y; dst[6]=(_Float16)f1.z; dst[7]=(_Float16)f1.w;
    }
    { // stage B (weights, [n][k]): 128 x 32 f32 -> f16 (16 elems / thread)
      const int r  = tid >> 1;
      const int kk = (tid & 1) * 16;
      const float* src = W + (size_t)(n0 + r) * K + k0 + kk;
      _Float16* dst = Bsh + r * GLD + kk;
#pragma unroll
      for (int q = 0; q < 4; ++q) {
        float4 f = *(const float4*)(src + q * 4);
        dst[q*4+0]=(_Float16)f.x; dst[q*4+1]=(_Float16)f.y;
        dst[q*4+2]=(_Float16)f.z; dst[q*4+3]=(_Float16)f.w;
      }
    }
    __syncthreads();

    v16h a0 = load_a_frag(Ash, mw,      GLD, 0, lane);
    v16h a1 = load_a_frag(Ash, mw + 16, GLD, 0, lane);
    v16h b0 = load_b_frag(Bsh, nw,      GLD, 0, lane);
    v16h b1 = load_b_frag(Bsh, nw + 16, GLD, 0, lane);
    acc[0][0] = wmma_f32_f16(a0, b0, acc[0][0]);
    acc[0][1] = wmma_f32_f16(a0, b1, acc[0][1]);
    acc[1][0] = wmma_f32_f16(a1, b0, acc[1][0]);
    acc[1][1] = wmma_f32_f16(a1, b1, acc[1][1]);
  }

  const int rb = (lane >> 4) * 8;      // C layout: M = i + 8*(lane>=16)
  const int cl = lane & 15;            //           N = lane & 15
#pragma unroll
  for (int mt = 0; mt < 2; ++mt)
#pragma unroll
    for (int nt = 0; nt < 2; ++nt) {
      const int colg = n0 + nw + nt * 16 + cl;
      const float bias = bih[colg] + bhh[colg];
#pragma unroll
      for (int i = 0; i < 8; ++i) {
        const int rowg = m0 + mw + mt * 16 + rb + i;
        out[(size_t)rowg * HID + colg] = acc[mt][nt][i] + bias;
      }
    }
}

// ---------------------------------------------------------------------------
// Persistent recurrence: h_t = tanh(xw_t + h_{t-1} @ W_hh^T), 4 WGs, each owns
// a 128-column slice of H with its W_hh^T slice resident in LDS (f16).
// h ping-pongs through global f16 buffers; per-step staging of h into LDS is
// done by the Tensor Data Mover (one tensor_load_to_lds per WG, HW-padded rows)
// when available. Grid barrier (atomic counter) per step.
// ---------------------------------------------------------------------------
__global__ __launch_bounds__(256)
void rnn_recurrence(const float* __restrict__ xw, const float* __restrict__ Whh,
                    float* __restrict__ y, _Float16* __restrict__ hbuf,
                    unsigned* __restrict__ ctr) {
  extern __shared__ char smem[];
  _Float16* Hsh = (_Float16*)smem;                                           // [BATCH][HLD]
  _Float16* Wsh = (_Float16*)(smem + (size_t)BATCH * HLD * sizeof(_Float16)); // [RNS][WLD]

  const int tid  = threadIdx.x;
  const int lane = tid & 31;
  const int wave = tid >> 5;
  const int n0 = blockIdx.x * RNS;
  const int mw = (wave & 1) * 32;      // 2 waves cover BATCH=64 rows
  const int nw = (wave >> 1) * 32;     // 4 waves cover 128 local cols
  const int rb = (lane >> 4) * 8;
  const int cl = lane & 15;

  // One-time: cache this WG's W_hh slice in LDS as f16, layout [n_local][k]
  for (int idx = tid; idx < RNS * (HID / 4); idx += 256) {
    const int n  = idx >> 7;           // HID/4 = 128 float4 per row
    const int k4 = (idx & 127) * 4;
    float4 f = *(const float4*)(Whh + (size_t)(n0 + n) * HID + k4);
    _Float16* d = Wsh + n * WLD + k4;
    d[0]=(_Float16)f.x; d[1]=(_Float16)f.y; d[2]=(_Float16)f.z; d[3]=(_Float16)f.w;
  }

  unsigned round = 0;

  for (int t = 0; t < SEQ; ++t) {
    const float* xwt = xw + (size_t)t * BATCH * HID;
    float*       yt  = y  + (size_t)t * BATCH * HID;
    _Float16*       hcur  = hbuf + (size_t)(t & 1)       * BATCH * HID;
    const _Float16* hprev = hbuf + (size_t)((t + 1) & 1) * BATCH * HID;

    if (t == 0) {                      // h_{-1} == 0: no matmul contribution
      for (int idx = tid; idx < BATCH * RNS; idx += 256) {
        const int b  = idx >> 7;
        const int nl = idx & 127;
        const float v = tanhf(xwt[b * HID + n0 + nl]);
        yt[b * HID + n0 + nl] = v;
        hcur[b * HID + n0 + nl] = (_Float16)v;
      }
    } else {
#if HAVE_TDM
      // ---- TDM: DMA h_{t-1} (64x512 f16) global -> LDS, HW-padding each
      //      512-f16 row (256 dwords) by 4 dwords => row stride HLD=520. ----
      if (wave == 0) {
        const unsigned long long ga = (unsigned long long)(uintptr_t)hprev;
        const unsigned hoff = (unsigned)(uintptr_t)Hsh;  // LDS byte offset (low 32 bits)
        u32x4 g0 = { 1u,                                  // count=1, user D#
                     hoff,                                // lds_addr
                     (unsigned)ga,                        // global_addr[31:0]
                     (unsigned)((ga >> 32) & 0x1FFFFFFu) | (2u << 30) }; // addr[56:32], type=2
        i32x8 g1 = { (int)((1u << 16)                     // data_size = 2 bytes
                         | (1u << 20)                     // pad_enable
                         | (7u << 22)                     // pad_interval: 256 dwords
                         | (3u << 25)),                   // pad_amount: 4 dwords
                     (int)(512u << 16),                   // tensor_dim0 = 512 (lo16 in [31:16])
                     (int)(64u  << 16),                   // dim0 hi=0 | tensor_dim1 = 64
                     (int)(512u << 16),                   // dim1 hi=0 | tile_dim0 = 512
                     (int)64,                             // tile_dim1 = 64, tile_dim2 = 0
                     (int)512,                            // tensor_dim0_stride = 512
                     0, 0 };                              // dim1_stride = 0 (2D)
        i32x4 gz  = { 0, 0, 0, 0 };
        i32x8 gz8 = { 0, 0, 0, 0, 0, 0, 0, 0 };
        __builtin_amdgcn_tensor_load_to_lds(g0, g1, gz, gz, gz8, 0);
      }
#else
      // fallback: cooperative VGPR round-trip copy
      for (int idx = tid; idx < BATCH * (HID / 8); idx += 256) {
        const int r  = idx >> 6;
        const int c8 = (idx & 63) * 8;
        *(v8h*)(Hsh + r * HLD + c8) = *(const v8h*)(hprev + r * HID + c8);
      }
#endif

      // overlap with the DMA: init accumulators from xw_t (L2 loads)
      v8f acc[2][2];
#pragma unroll
      for (int mt = 0; mt < 2; ++mt)
#pragma unroll
        for (int nt = 0; nt < 2; ++nt) {
          const int colg = n0 + nw + nt * 16 + cl;
          const float* src = xwt + (size_t)(mw + mt * 16 + rb) * HID + colg;
#pragma unroll
          for (int i = 0; i < 8; ++i) acc[mt][nt][i] = src[(size_t)i * HID];
        }

      // prefetch next step's xw slice for this WG (hidden under WMMA loop)
      if (t + 1 < SEQ) {
        const float* xwn = xwt + (size_t)BATCH * HID;
        __builtin_prefetch(xwn + (size_t)(tid >> 2) * HID + n0 + (tid & 3) * 32, 0, 1);
      }

#if HAVE_TDM
      if (wave == 0) __builtin_amdgcn_s_wait_tensorcnt(0);
#endif
      __syncthreads();                 // Hsh published to all waves

#pragma unroll
      for (int k0 = 0; k0 < HID; k0 += 32) {
        v16h a0 = load_a_frag(Hsh, mw,      HLD, k0, lane);
        v16h a1 = load_a_frag(Hsh, mw + 16, HLD, k0, lane);
        v16h b0 = load_b_frag(Wsh, nw,      WLD, k0, lane);
        v16h b1 = load_b_frag(Wsh, nw + 16, WLD, k0, lane);
        acc[0][0] = wmma_f32_f16(a0, b0, acc[0][0]);
        acc[0][1] = wmma_f32_f16(a0, b1, acc[0][1]);
        acc[1][0] = wmma_f32_f16(a1, b0, acc[1][0]);
        acc[1][1] = wmma_f32_f16(a1, b1, acc[1][1]);
      }

#pragma unroll
      for (int mt = 0; mt < 2; ++mt)
#pragma unroll
        for (int nt = 0; nt < 2; ++nt) {
          const int colg = n0 + nw + nt * 16 + cl;
#pragma unroll
          for (int i = 0; i < 8; ++i) {
            const int rowg = mw + mt * 16 + rb + i;
            const float v = tanhf(acc[mt][nt][i]);
            yt[(size_t)rowg * HID + colg] = v;
            hcur[(size_t)rowg * HID + colg] = (_Float16)v;
          }
        }
    }

    // ---- grid barrier across the RWG persistent workgroups ----
    __syncthreads();
    round += 1;
    if (tid == 0) {
      __threadfence();
      atomicAdd(ctr, 1u);
      const unsigned target = round * RWG;
      while (__hip_atomic_load(ctr, __ATOMIC_RELAXED, __HIP_MEMORY_SCOPE_AGENT) < target)
        __builtin_amdgcn_s_sleep(1);
      __threadfence();
    }
    __syncthreads();
  }
}

__global__ void rnn_zero_ctr(unsigned* c) {
  if (threadIdx.x < 8) c[threadIdx.x] = 0u;
}

// ---------------------------------------------------------------------------
extern "C" void kernel_launch(void* const* d_in, const int* in_sizes, int n_in,
                              void* d_out, int out_size, void* d_ws, size_t ws_size,
                              hipStream_t stream) {
  (void)in_sizes; (void)n_in; (void)out_size; (void)ws_size;
  const float* x       = (const float*)d_in[0];  // [S,B,IN]
  const float* w_ih_f  = (const float*)d_in[1];  // [H,IN]
  const float* w_ih_r  = (const float*)d_in[2];  // [2,H,H]
  const float* w_hh    = (const float*)d_in[3];  // [3,H,H]
  const float* b_ih    = (const float*)d_in[4];  // [3,H]
  const float* b_hh    = (const float*)d_in[5];  // [3,H]
  float* y = (float*)d_out;                      // [S,B,H]

  // workspace layout
  float* xw = (float*)d_ws;                                    // 64 MB
  char* p = (char*)d_ws + (size_t)SEQ * BATCH * HID * sizeof(float);
  _Float16* hbuf = (_Float16*)p;                               // 2 x [B,H] f16
  unsigned* ctr  = (unsigned*)(p + (size_t)2 * BATCH * HID * sizeof(_Float16));

  const int rec_lds = (int)(((size_t)RNS * WLD + (size_t)BATCH * HLD) * sizeof(_Float16));
  (void)hipFuncSetAttribute((const void*)rnn_recurrence,
                            hipFuncAttributeMaxDynamicSharedMemorySize, rec_lds);

  rnn_zero_ctr<<<1, 64, 0, stream>>>(ctr);

  const dim3 ggrid(SEQ * BATCH / GBM, HID / GBN);

  // layer 0
  rnn_xw_gemm<<<ggrid, 256, 0, stream>>>(x, w_ih_f, b_ih, b_hh, xw, INDIM);
  rnn_recurrence<<<RWG, 256, rec_lds, stream>>>(xw, w_hh, y, hbuf, ctr + 0);
  // layer 1
  rnn_xw_gemm<<<ggrid, 256, 0, stream>>>(y, w_ih_r, b_ih + HID, b_hh + HID, xw, HID);
  rnn_recurrence<<<RWG, 256, rec_lds, stream>>>(xw, w_hh + (size_t)HID * HID, y, hbuf, ctr + 1);
  // layer 2
  rnn_xw_gemm<<<ggrid, 256, 0, stream>>>(y, w_ih_r + (size_t)HID * HID,
                                         b_ih + 2 * HID, b_hh + 2 * HID, xw, HID);
  rnn_recurrence<<<RWG, 256, rec_lds, stream>>>(xw, w_hh + (size_t)2 * HID * HID, y, hbuf, ctr + 2);
}
